// QINCoInferenceEncoder_62775241998557
// MI455X (gfx1250) — matrix operand order repaired and checked by reading, not compiled, not run on hardware
//
#include <hip/hip_runtime.h>
#include <hip/hip_bf16.h>

// ---------------------------------------------------------------------------
// QINCo inference encoder for MI455X (gfx1250).
// MLP chain (~1.56 TFLOP) runs on v_wmma_f32_16x16x32_bf16; weights are
// pre-swizzled into per-lane WMMA B-fragment layout; activations are staged in
// LDS bf16 with padded stride so A-fragments load as two ds_load_b128.
// ---------------------------------------------------------------------------

typedef __attribute__((ext_vector_type(16))) __bf16 v16bf;
typedef __attribute__((ext_vector_type(8)))  float  v8f;

#define BATCH 2048
#define DIM   128
#define KCB   256
#define HID   256
#define BEAM  16
#define ACAND 8
#define NSTEP 7
#define MTOT  8
#define LDSW  392      // padded activation row stride in halves (384 + 8)

// swizzled-weight half offsets within one step
#define OFF_WIN   0
#define OFF_WCAT  32768
#define OFF_W1_0  131072
#define OFF_W2_0  196608
#define OFF_W1_1  262144
#define OFF_W2_1  327680
#define OFF_WOUT  393216
#define STEP_HALVES 425984
#define FRAGS_PER_STEP 832

#define FLT_BIG 3.402823466e38f

// ---------------------------------------------------------------------------
// Weight prep: f32 row-major [K][N] -> bf16 WMMA B-fragments.
// Fragment (kt,nt): lane l holds 16 halves: W[kt*32 + (l>>4)*16 + j][nt*16 + (l&15)]
// stored contiguously (32 B per lane) so the GEMM loads each fragment with
// two global b128 loads per lane.
// ---------------------------------------------------------------------------
__global__ void k_prep_weights(const float* __restrict__ Win,
                               const float* __restrict__ Wcat,
                               const float* __restrict__ W1,
                               const float* __restrict__ W2,
                               const float* __restrict__ Wout,
                               __bf16* __restrict__ wsW)
{
  int fid  = blockIdx.x % FRAGS_PER_STEP;
  int s    = blockIdx.x / FRAGS_PER_STEP;
  int lane = threadIdx.x;
  const float* src; int N; size_t off; int f;
  if      (fid <  64) { src = Win  + (size_t)s*DIM*HID;       N = HID; off = OFF_WIN;  f = fid;       }
  else if (fid < 256) { src = Wcat + (size_t)s*(HID+DIM)*HID; N = HID; off = OFF_WCAT; f = fid - 64;  }
  else if (fid < 384) { src = W1   + ((size_t)s*2+0)*HID*HID; N = HID; off = OFF_W1_0; f = fid - 256; }
  else if (fid < 512) { src = W2   + ((size_t)s*2+0)*HID*HID; N = HID; off = OFF_W2_0; f = fid - 384; }
  else if (fid < 640) { src = W1   + ((size_t)s*2+1)*HID*HID; N = HID; off = OFF_W1_1; f = fid - 512; }
  else if (fid < 768) { src = W2   + ((size_t)s*2+1)*HID*HID; N = HID; off = OFF_W2_1; f = fid - 640; }
  else                { src = Wout + (size_t)s*HID*DIM;       N = DIM; off = OFF_WOUT; f = fid - 768; }
  int NT = N >> 4;
  int kt = f / NT, nt = f % NT;
  __bf16* dst = wsW + (size_t)s*STEP_HALVES + off + (size_t)f*512 + lane*16;
  int col  = nt*16 + (lane & 15);
  int krow = kt*32 + ((lane >> 4) << 4);
  for (int j = 0; j < 16; ++j)
    dst[j] = (__bf16)src[(size_t)(krow + j)*N + col];
}

// ---------------------------------------------------------------------------
// Step 0: d = sum c*(c - 2x) over codebook[0]; iterative top-BEAM argmin.
// ---------------------------------------------------------------------------
__global__ void k_step0(const float* __restrict__ x, const float* __restrict__ cb0,
                        float* __restrict__ xhatOut, int* __restrict__ codesOut)
{
  __shared__ float xs[DIM];
  __shared__ float dv[KCB];
  __shared__ float rv[KCB];
  __shared__ int   ri[KCB];
  const int b = blockIdx.x, t = threadIdx.x;
  if (t < DIM) xs[t] = x[(size_t)b*DIM + t];
  __syncthreads();
  const float* c = cb0 + (size_t)t*DIM;
  float acc = 0.f;
  for (int d = 0; d < DIM; ++d) { float cv = c[d]; acc += cv*(cv - 2.f*xs[d]); }
  dv[t] = acc;
  __syncthreads();
  for (int r = 0; r < BEAM; ++r) {
    rv[t] = dv[t]; ri[t] = t;
    __syncthreads();
    for (int off = KCB/2; off > 0; off >>= 1) {
      if (t < off && rv[t+off] < rv[t]) { rv[t] = rv[t+off]; ri[t] = ri[t+off]; }
      __syncthreads();
    }
    int win = ri[0];
    if (t == 0) { codesOut[b*BEAM + r] = win; dv[win] = FLT_BIG; }
    if (t < DIM) xhatOut[((size_t)b*BEAM + r)*DIM + t] = cb0[(size_t)win*DIM + t];
    __syncthreads();
  }
}

// ---------------------------------------------------------------------------
// Shortlist: dists of residuals (x - xhat[f]) vs codebook_rq[s]; top-A per f.
// ---------------------------------------------------------------------------
__global__ void k_shortlist(const float* __restrict__ x, const float* __restrict__ cbrq_s,
                            const float* __restrict__ xhatIn, int* __restrict__ topC)
{
  __shared__ float xs[DIM];
  __shared__ float xh[BEAM][DIM];
  __shared__ float dist[BEAM][KCB];
  const int b = blockIdx.x, t = threadIdx.x;
  if (t < DIM) xs[t] = x[(size_t)b*DIM + t];
  for (int idx = t; idx < BEAM*DIM; idx += 256)
    xh[idx >> 7][idx & 127] = xhatIn[(size_t)b*BEAM*DIM + idx];
  __syncthreads();
  for (int idx = t; idx < BEAM*KCB; idx += 256) {
    int f = idx >> 8, k = idx & 255;
    const float* c  = cbrq_s + (size_t)k*DIM;
    const float* xf = xh[f];
    float acc = 0.f;
    for (int d = 0; d < DIM; ++d) { float cv = c[d]; acc += cv*(cv - 2.f*(xs[d] - xf[d])); }
    dist[f][k] = acc;
  }
  __syncthreads();
  const int w = t >> 5, lane = t & 31;   // wave w handles f = 2w, 2w+1
  for (int r = 0; r < ACAND; ++r) {
    for (int ff = 0; ff < 2; ++ff) {
      int f = 2*w + ff;
      float best = FLT_BIG; int bi = 0;
      for (int k = lane; k < KCB; k += 32) {
        float v = dist[f][k];
        if (v < best) { best = v; bi = k; }
      }
      for (int o = 16; o > 0; o >>= 1) {
        float ov = __shfl_xor(best, o);
        int   oi = __shfl_xor(bi, o);
        if (ov < best) { best = ov; bi = oi; }
      }
      if (lane == 0) { topC[((size_t)b*BEAM + f)*ACAND + r] = bi; dist[f][bi] = FLT_BIG; }
    }
    __syncthreads();
  }
}

// ---------------------------------------------------------------------------
// WMMA MLP engine.
// ---------------------------------------------------------------------------
__device__ __forceinline__ void wave_lds_sync() {
  asm volatile("s_wait_dscnt 0" ::: "memory");
}

// A fragment (16x32 bf16): lane l (row l&15) holds
// K = kb..kb+15 with kb = kt*32 + (l>>4)*16. Two b128 LDS loads.
__device__ __forceinline__ v16bf lda_frag(const __bf16* aBuf, int lane, int kt) {
  const __bf16* p = aBuf + (size_t)(lane & 15)*LDSW + kt*32 + ((lane >> 4) << 4);
  v16bf a;
  ((uint4*)&a)[0] = *(const uint4*)(p);
  ((uint4*)&a)[1] = *(const uint4*)(p + 8);
  return a;
}

// 16 rows x (NT*16) cols = act(16 x KT*32) @ W + bias [+res] [relu] -> bf16 LDS
__device__ __forceinline__ void gemm_layer(const __bf16* aBuf, const v16bf* w,
                                           int KT, int NT, const float* bias,
                                           __bf16* outBuf, int outCol,
                                           const __bf16* resBuf, bool doRelu, int lane)
{
  const int col   = lane & 15;
  const int rbase = (lane >> 4) << 3;
  for (int nt = 0; nt < NT; ++nt) {
    v8f acc = {};
    for (int kt = 0; kt < KT; ++kt) {
      v16bf a  = lda_frag(aBuf, lane, kt);
      v16bf bf = w[(size_t)(kt*NT + nt)*32 + lane];
      acc = __builtin_amdgcn_wmma_f32_16x16x32_bf16(false, a, false, bf,
                                                    (short)0, acc, false, false);
    }
    float bv = bias[nt*16 + col];
    for (int r = 0; r < 8; ++r) {
      int m = rbase + r;
      float v = acc[r] + bv;
      if (resBuf) v += (float)resBuf[(size_t)m*LDSW + outCol + nt*16 + col];
      if (doRelu) v = fmaxf(v, 0.f);
      outBuf[(size_t)m*LDSW + outCol + nt*16 + col] = (__bf16)v;
    }
  }
}

// One workgroup = 4 waves = 64 candidate rows of batch element b (half of 128).
// Each wave owns a 16-row tile and runs the full layer chain wave-synchronously.
__global__ void __launch_bounds__(128)
k_mlp(const float* __restrict__ xhatIn, const int* __restrict__ topC,
      const float* __restrict__ cbStep, const __bf16* __restrict__ wstep,
      const float* __restrict__ b_in, const float* __restrict__ b_cat,
      const float* __restrict__ b1, const float* __restrict__ b2,
      const float* __restrict__ b_out, float* __restrict__ cand)
{
  __shared__ __align__(16) __bf16 act[4][2][16*LDSW];   // double buffer per wave
  __shared__ float xh[8][DIM];
  __shared__ float cwres[64][DIM];
  __shared__ int   codes_sh[64];

  const int t = threadIdx.x;
  const int w = t >> 5, lane = t & 31;
  const int b = blockIdx.x >> 1, half = blockIdx.x & 1;

  for (int idx = t; idx < 8*DIM; idx += 128)
    xh[idx >> 7][idx & 127] =
        xhatIn[((size_t)b*BEAM + half*8 + (idx >> 7))*DIM + (idx & 127)];
  if (t < 64) codes_sh[t] = topC[(size_t)b*(BEAM*ACAND) + half*64 + t];
  __syncthreads();

  __bf16* bufA = act[w][0];
  __bf16* bufB = act[w][1];

  // gather this wave's 16 codewords: residual copy (f32) + bf16 A-activations
  for (int idx = lane; idx < 16*DIM; idx += 32) {
    int r = idx >> 7, d = idx & 127;
    int code = codes_sh[w*16 + r];
    float cv = cbStep[(size_t)code*DIM + d];
    cwres[w*16 + r][d] = cv;
    bufA[(size_t)r*LDSW + d] = (__bf16)cv;
  }
  wave_lds_sync();

  const v16bf* W = (const v16bf*)wstep;

  // h = cw @ W_in + b_in                          (128 -> 256)
  gemm_layer(bufA, W + OFF_WIN/16, 4, 16, b_in, bufB, 0, nullptr, false, lane);
  // append xhat (bf16) at cols 256..383
  for (int idx = lane; idx < 16*DIM; idx += 32) {
    int r = idx >> 7, d = idx & 127;
    int fl = 2*w + (r >> 3);
    bufB[(size_t)r*LDSW + HID + d] = (__bf16)xh[fl][d];
  }
  wave_lds_sync();
  // h = [h, xhat] @ W_cat + b_cat                 (384 -> 256)
  gemm_layer(bufB, W + OFF_WCAT/16, 12, 16, b_cat, bufA, 0, nullptr, false, lane);
  wave_lds_sync();
  // residual block 0
  gemm_layer(bufA, W + OFF_W1_0/16, 8, 16, b1,       bufB, 0, nullptr, true,  lane);
  wave_lds_sync();
  gemm_layer(bufB, W + OFF_W2_0/16, 8, 16, b2,       bufA, 0, bufA,    false, lane);
  wave_lds_sync();
  // residual block 1
  gemm_layer(bufA, W + OFF_W1_1/16, 8, 16, b1 + HID, bufB, 0, nullptr, true,  lane);
  wave_lds_sync();
  gemm_layer(bufB, W + OFF_W2_1/16, 8, 16, b2 + HID, bufA, 0, bufA,    false, lane);
  wave_lds_sync();

  // cand = h @ W_out + b_out + codeword + xhat    (256 -> 128), f32 to global
  {
    const v16bf* Wo = W + OFF_WOUT/16;
    const int col = lane & 15, rbase = (lane >> 4) << 3;
    for (int nt = 0; nt < 8; ++nt) {
      v8f acc = {};
      for (int kt = 0; kt < 8; ++kt) {
        v16bf a  = lda_frag(bufA, lane, kt);
        v16bf bf = Wo[(size_t)(kt*8 + nt)*32 + lane];
        acc = __builtin_amdgcn_wmma_f32_16x16x32_bf16(false, a, false, bf,
                                                      (short)0, acc, false, false);
      }
      float bv = b_out[nt*16 + col];
      for (int r = 0; r < 8; ++r) {
        int m  = rbase + r;
        int g  = half*64 + w*16 + m;        // candidate row within batch elem
        int fl = 2*w + (m >> 3);
        int d  = nt*16 + col;
        float v = acc[r] + bv + cwres[w*16 + m][d] + xh[fl][d];
        cand[((size_t)b*128 + g)*DIM + d] = v;
      }
    }
  }
}

// ---------------------------------------------------------------------------
// Re-score candidates, prune beam, reorder code lists, emit final outputs.
// ---------------------------------------------------------------------------
__global__ void k_select(const float* __restrict__ x, const float* __restrict__ cand,
                         const int* __restrict__ topC, const int* __restrict__ codesSrc,
                         int* __restrict__ codesDst, float* __restrict__ xhatDst,
                         int nLists, int Fout, float* __restrict__ outFinal)
{
  __shared__ float xs[DIM];
  __shared__ float d2[128];
  __shared__ float rv[128];
  __shared__ int   ri[128];
  __shared__ int   sel[BEAM];
  const int b = blockIdx.x, t = threadIdx.x;
  xs[t] = x[(size_t)b*DIM + t];
  __syncthreads();
  const float* cb = cand + (size_t)b*128*DIM;
  const float* cc = cb + (size_t)t*DIM;
  float acc = 0.f;
  for (int d = 0; d < DIM; ++d) { float v = cc[d]; acc += v*(v - 2.f*xs[d]); }
  d2[t] = acc;
  __syncthreads();
  for (int r = 0; r < Fout; ++r) {
    rv[t] = d2[t]; ri[t] = t;
    __syncthreads();
    for (int off = 64; off > 0; off >>= 1) {
      if (t < off && rv[t+off] < rv[t]) { rv[t] = rv[t+off]; ri[t] = ri[t+off]; }
      __syncthreads();
    }
    if (t == 0) { sel[r] = ri[0]; d2[ri[0]] = FLT_BIG; }
    __syncthreads();
  }
  // reorder existing code lists by parent + append new codes
  for (int idx = t; idx < (nLists + 1)*Fout; idx += 128) {
    int l = idx / Fout, r = idx % Fout;
    int j = sel[r], parent = j >> 3;
    int v = (l < nLists) ? codesSrc[(size_t)l*(BATCH*BEAM) + b*BEAM + parent]
                         : topC[(size_t)b*(BEAM*ACAND) + j];
    codesDst[(size_t)l*(BATCH*BEAM) + b*BEAM + r] = v;
  }
  // new xhat = selected candidates
  for (int idx = t; idx < Fout*DIM; idx += 128) {
    int r = idx >> 7, d = idx & 127;
    xhatDst[((size_t)b*BEAM + r)*DIM + d] = cb[(size_t)sel[r]*DIM + d];
  }
  if (outFinal) {                         // last step: Fout == 1
    __syncthreads();
    if (t < MTOT)
      outFinal[(size_t)t*BATCH + b] =
          (float)codesDst[(size_t)t*(BATCH*BEAM) + b*BEAM];
    outFinal[(size_t)MTOT*BATCH + (size_t)b*DIM + t] = cb[(size_t)sel[0]*DIM + t];
  }
}

// ---------------------------------------------------------------------------
// Host launcher. Workspace layout (bytes):
//   [0)          swizzled bf16 weights          5,963,776
//   [5963776)    xhat ping (B,16,128 f32)      16,777,216
//   [22740992)   xhat pong                     16,777,216
//   [39518208)   codes ping (M,B,16 i32)        2,097,152
//   [41615360)   codes pong                     2,097,152
//   [43712512)   topC (B,16,8 i32)              1,048,576
//   [44761088)   cand (B,128,128 f32)         134,217,728   -> ~171 MB total
// ---------------------------------------------------------------------------
extern "C" void kernel_launch(void* const* d_in, const int* in_sizes, int n_in,
                              void* d_out, int out_size, void* d_ws, size_t ws_size,
                              hipStream_t stream)
{
  const float* x    = (const float*)d_in[0];
  const float* cb   = (const float*)d_in[1];
  const float* cbrq = (const float*)d_in[2];
  const float* Win  = (const float*)d_in[3];
  const float* bin  = (const float*)d_in[4];
  const float* Wcat = (const float*)d_in[5];
  const float* bcat = (const float*)d_in[6];
  const float* W1   = (const float*)d_in[7];
  const float* b1   = (const float*)d_in[8];
  const float* W2   = (const float*)d_in[9];
  const float* b2   = (const float*)d_in[10];
  const float* Wout = (const float*)d_in[11];
  const float* bout = (const float*)d_in[12];

  char* ws = (char*)d_ws;
  __bf16* wsW   = (__bf16*)(ws);
  float* xhatA  = (float*)(ws + 5963776);
  float* xhatB  = (float*)(ws + 22740992);
  int*   codesA = (int*)  (ws + 39518208);
  int*   codesB = (int*)  (ws + 41615360);
  int*   topC   = (int*)  (ws + 43712512);
  float* cand   = (float*)(ws + 44761088);

  k_prep_weights<<<NSTEP*FRAGS_PER_STEP, 32, 0, stream>>>(Win, Wcat, W1, W2, Wout, wsW);
  k_step0<<<BATCH, 256, 0, stream>>>(x, cb, xhatA, codesA);

  for (int s = 0; s < NSTEP; ++s) {
    float* xhatSrc = (s & 1) ? xhatB : xhatA;
    float* xhatDst = (s & 1) ? xhatA : xhatB;
    int* codesSrc  = (s & 1) ? codesB : codesA;
    int* codesDst  = (s & 1) ? codesA : codesB;
    int Fout = (s < NSTEP - 1) ? BEAM : 1;

    k_shortlist<<<BATCH, 256, 0, stream>>>(x, cbrq + (size_t)s*KCB*DIM, xhatSrc, topC);
    k_mlp<<<BATCH*2, 128, 0, stream>>>(xhatSrc, topC,
        cb + (size_t)(s + 1)*KCB*DIM, wsW + (size_t)s*STEP_HALVES,
        bin + (size_t)s*HID, bcat + (size_t)s*HID,
        b1 + (size_t)s*2*HID, b2 + (size_t)s*2*HID,
        bout + (size_t)s*DIM, cand);
    k_select<<<BATCH, 128, 0, stream>>>(x, cand, topC, codesSrc, codesDst, xhatDst,
        s + 1, Fout, (s == NSTEP - 1) ? (float*)d_out : nullptr);
  }
  (void)in_sizes; (void)n_in; (void)out_size; (void)ws_size;
}
